// MPSFractalBlock_80994493268374
// MI455X (gfx1250) — matrix operand order repaired and checked by reading, compiled, not run on hardware
//
#include <hip/hip_runtime.h>

// ======================================================================
// MPS/TT transformer block for gfx1250 (MI455X), bf16 WMMA + fp32 residuals.
// Workspace layout (bytes, total 88,080,384):
//   [0        ) tmpA (16.8MB f32)  -> later X1 (residual stream, fp32)
//   [16777216 ) tmpB (16.8MB f32)  -> later gate/silu buffer (bf16)
//   [33554432 ) Wqkv bf16 3072x1024
//   [39845888 ) Wo   bf16 1024x1024
//   [41943040 ) Wg   bf16 2048x1024
//   [46137344 ) Wu   bf16 2048x1024
//   [50331648 ) Wd   bf16 1024x2048
//   [54525952 ) Hbuf bf16 4096x1024 (h -> attn_out -> h2)
//   [62914560 ) QKV  bf16 4096x3072 -> later up buffer (bf16)
// ======================================================================

typedef __bf16 v16bf __attribute__((ext_vector_type(16)));
typedef float  v8f   __attribute__((ext_vector_type(8)));

#define WMMA_BF16(a, b, c) \
  __builtin_amdgcn_wmma_f32_16x16x32_bf16(false, (a), false, (b), (short)0, (c), false, false)

union FragB { unsigned u[8]; v16bf v; };

static __device__ __forceinline__ unsigned short f2bf(float f) {
  unsigned u = __builtin_bit_cast(unsigned, f);
  u += 0x7FFFu + ((u >> 16) & 1u);           // round-to-nearest-even
  return (unsigned short)(u >> 16);
}
static __device__ __forceinline__ float bf2f(unsigned short h) {
  unsigned u = ((unsigned)h) << 16;
  return __builtin_bit_cast(float, u);
}

// ---------------------------------------------------------------- TT build
// Tout[(m*pq + r)*br + e] = sum_d Tin[m*16+d] * core[(d*pq + r)*br + e]
__global__ void tt_contract(const float* __restrict__ Tin, const float* __restrict__ core,
                            float* __restrict__ Tout, int M, int pq, int br) {
  int i = blockIdx.x * blockDim.x + threadIdx.x;
  int total = M * pq * br;
  if (i >= total) return;
  int e = i % br;
  int r = (i / br) % pq;
  int m = i / (br * pq);
  float acc = 0.f;
#pragma unroll
  for (int d = 0; d < 16; ++d)
    acc += Tin[m * 16 + d] * core[(d * pq + r) * br + e];
  Tout[i] = acc;
}

// W[r, c] (row-major out_f x in_f) from T5 flat with interleaved digits
// p1 q1 p2 q2 ... p5 q5; r = (q1..q5) base-q, c = (p1..p5) base-p.
__global__ void tt_materialize(const float* __restrict__ T5, unsigned short* __restrict__ W,
                               int p, int q, int out_f, int in_f) {
  int i = blockIdx.x * blockDim.x + threadIdx.x;
  if (i >= out_f * in_f) return;
  int r = i / in_f, c = i % in_f;
  int ppow = p * p * p * p;
  int qpow = q * q * q * q;
  int idx = 0, rr = r, cc = c;
#pragma unroll
  for (int s = 0; s < 5; ++s) {
    int qd = rr / qpow; rr -= qd * qpow;
    int pd = cc / ppow; cc -= pd * ppow;
    idx = (idx * p + pd) * q + qd;
    qpow /= q; ppow /= p;
  }
  W[i] = f2bf(T5[idx]);
}

// ---------------------------------------------------------------- RMSNorm
__global__ __launch_bounds__(256) void rmsnorm_kernel(
    const float* __restrict__ x, const float* __restrict__ w,
    unsigned short* __restrict__ out, int D) {
  __shared__ float red[8];
  int row = blockIdx.x;
  const float* xr = x + (size_t)row * D;
  float ss = 0.f;
  for (int j = threadIdx.x; j < D; j += 256) { float v = xr[j]; ss += v * v; }
#pragma unroll
  for (int m = 16; m >= 1; m >>= 1) ss += __shfl_xor(ss, m, 32);
  if ((threadIdx.x & 31) == 0) red[threadIdx.x >> 5] = ss;
  __syncthreads();
  float tot = 0.f;
#pragma unroll
  for (int k = 0; k < 8; ++k) tot += red[k];
  float scale = rsqrtf(tot / (float)D + 1.1920929e-07f);
  unsigned short* orow = out + (size_t)row * D;
  for (int j = threadIdx.x; j < D; j += 256)
    orow[j] = f2bf(xr[j] * scale * w[j]);
}

// ---------------------------------------------------------------- NT GEMM
// C[M,N] = A[M,K] * W[N,K]^T (+Res), bf16 inputs, fp32 accumulate.
// 256 threads, 8 waves in 2(M)x4(N); wave computes 64x32 via 8 WMMA/K-step.
template <int OUT_BF16, int RESIDUAL>
__global__ __launch_bounds__(256) void gemm_nt_kernel(
    const unsigned short* __restrict__ A, const unsigned short* __restrict__ W,
    const float* __restrict__ Res, void* __restrict__ Cout, int M, int N, int K) {
  __shared__ unsigned As[128 * 17];   // 128 rows x 32 bf16 (16 u32 + pad)
  __shared__ unsigned Ws[128 * 17];
  const int t = threadIdx.x;
  const int lane = t & 31, wave = t >> 5;
  const int hi = lane >> 4, l16 = lane & 15;
  const int wm = wave >> 2, wn = wave & 3;
  const int bm = blockIdx.y * 128, bn = blockIdx.x * 128;

  v8f acc[4][2];
#pragma unroll
  for (int i = 0; i < 4; ++i)
#pragma unroll
    for (int j = 0; j < 2; ++j)
      acc[i][j] = (v8f){0.f, 0.f, 0.f, 0.f, 0.f, 0.f, 0.f, 0.f};

  for (int k0 = 0; k0 < K; k0 += 32) {
    __syncthreads();
#pragma unroll
    for (int j = 0; j < 2; ++j) {            // 512 uint4 per tile, 2/thread
      int l = t + j * 256;
      int row = l >> 2, quad = l & 3;
      uint4 va = *((const uint4*)(A + (size_t)(bm + row) * K + k0) + quad);
      unsigned* da = As + row * 17 + quad * 4;
      da[0] = va.x; da[1] = va.y; da[2] = va.z; da[3] = va.w;
      uint4 vw = *((const uint4*)(W + (size_t)(bn + row) * K + k0) + quad);
      unsigned* dw = Ws + row * 17 + quad * 4;
      dw[0] = vw.x; dw[1] = vw.y; dw[2] = vw.z; dw[3] = vw.w;
    }
    __syncthreads();
    v16bf bfr[2];
#pragma unroll
    for (int ni = 0; ni < 2; ++ni) {         // B frag: lane=N, K pairs linear
      const unsigned* p = Ws + (wn * 32 + ni * 16 + l16) * 17 + hi * 8;
      FragB r;
#pragma unroll
      for (int e = 0; e < 8; ++e) r.u[e] = p[e];
      bfr[ni] = r.v;
    }
#pragma unroll
    for (int mi = 0; mi < 4; ++mi) {         // A frag: lane=M, K pairs linear
      const unsigned* p = As + (wm * 64 + mi * 16 + l16) * 17;
      FragB r;
#pragma unroll
      for (int e = 0; e < 8; ++e) r.u[e] = p[hi * 8 + e];
#pragma unroll
      for (int ni = 0; ni < 2; ++ni)
        acc[mi][ni] = WMMA_BF16(r.v, bfr[ni], acc[mi][ni]);
    }
  }
#pragma unroll
  for (int mi = 0; mi < 4; ++mi)
#pragma unroll
    for (int ni = 0; ni < 2; ++ni)
#pragma unroll
      for (int e = 0; e < 8; ++e) {          // C: m = e + 8*hi, n = l16
        int gm = bm + wm * 64 + mi * 16 + e + 8 * hi;
        int gn = bn + wn * 32 + ni * 16 + l16;
        float val = acc[mi][ni][e];
        if (RESIDUAL) val += Res[(size_t)gm * N + gn];
        if (OUT_BF16) ((unsigned short*)Cout)[(size_t)gm * N + gn] = f2bf(val);
        else          ((float*)Cout)[(size_t)gm * N + gn] = val;
      }
}

// ---------------------------------------------------------------- Attention
// Flash-style causal attention. qkv layout (B,T,[3][H][64]) bf16.
// Block = (b,h,128 query rows); 8 waves x 16 rows. Out (B,T,H*64) bf16.
__global__ __launch_bounds__(256) void attention_kernel(
    const unsigned short* __restrict__ qkv, unsigned short* __restrict__ outp, int T) {
  __shared__ unsigned Ks[64 * 33];          // K tile rows (s), d-pairs packed
  __shared__ unsigned short VT[64 * 66];    // V transposed: rows d, cols s
  __shared__ unsigned Ps[8 * 16 * 33];      // per-wave P (16x64 bf16)
  const int t = threadIdx.x;
  const int lane = t & 31, wave = t >> 5;
  const int hi = lane >> 4, l16 = lane & 15;
  const int bid = blockIdx.x;
  const int qb = bid & 15;
  const int h = (bid >> 4) & 15;
  const int b = bid >> 8;
  const int qbase = qb * 128 + wave * 16;
  const unsigned short* base = qkv + (size_t)b * T * 3072;

  v16bf qf[2];                               // Q rows in registers (K=64)
  {
    const unsigned* qrow = (const unsigned*)(base + (size_t)(qbase + l16) * 3072 + h * 64);
#pragma unroll
    for (int j = 0; j < 2; ++j) {
      FragB r;
#pragma unroll
      for (int e = 0; e < 8; ++e)
        r.u[e] = qrow[j * 16 + hi * 8 + e];
      qf[j] = r.v;
    }
  }

  v8f oacc[4];
#pragma unroll
  for (int dn = 0; dn < 4; ++dn)
    oacc[dn] = (v8f){0.f, 0.f, 0.f, 0.f, 0.f, 0.f, 0.f, 0.f};
  float mrow[8], lrow[8];
#pragma unroll
  for (int e = 0; e < 8; ++e) { mrow[e] = -3.0e38f; lrow[e] = 0.f; }

  const int ktmax = qb * 2 + 1;
  for (int kt = 0; kt <= ktmax; ++kt) {
    __syncthreads();
#pragma unroll
    for (int j = 0; j < 2; ++j) {            // load K tile + transposed V tile
      int l = t + j * 256;
      int row = l >> 3, quad = l & 7;
      const unsigned short* krow = base + (size_t)(kt * 64 + row) * 3072 + 1024 + h * 64;
      uint4 vk = *((const uint4*)krow + quad);
      unsigned* dk = Ks + row * 33 + quad * 4;
      dk[0] = vk.x; dk[1] = vk.y; dk[2] = vk.z; dk[3] = vk.w;
      const unsigned short* vrow = base + (size_t)(kt * 64 + row) * 3072 + 2048 + h * 64;
      uint4 vv = *((const uint4*)vrow + quad);
      unsigned arr[4] = {vv.x, vv.y, vv.z, vv.w};
#pragma unroll
      for (int e2 = 0; e2 < 4; ++e2) {
        VT[(quad * 8 + 2 * e2 + 0) * 66 + row] = (unsigned short)(arr[e2] & 0xFFFFu);
        VT[(quad * 8 + 2 * e2 + 1) * 66 + row] = (unsigned short)(arr[e2] >> 16);
      }
    }
    __syncthreads();

    v8f sf[4];                               // S = Q*K^T (16x64)
#pragma unroll
    for (int nf = 0; nf < 4; ++nf) {
      v8f s = (v8f){0.f, 0.f, 0.f, 0.f, 0.f, 0.f, 0.f, 0.f};
#pragma unroll
      for (int ks = 0; ks < 2; ++ks) {
        const unsigned* p = Ks + (nf * 16 + l16) * 33 + ks * 16 + hi * 8;
        FragB r;
#pragma unroll
        for (int e = 0; e < 8; ++e) r.u[e] = p[e];
        s = WMMA_BF16(qf[ks], r.v, s);
      }
      sf[nf] = s;
    }

    float nm[8];
#pragma unroll
    for (int e = 0; e < 8; ++e) nm[e] = -3.0e38f;
#pragma unroll
    for (int nf = 0; nf < 4; ++nf)
#pragma unroll
      for (int e = 0; e < 8; ++e) {          // scale + causal mask
        float xv = sf[nf][e] * 0.125f;
        int qr = qbase + e + 8 * hi;
        int kc = kt * 64 + nf * 16 + l16;
        xv = (kc <= qr) ? xv : -3.0e38f;
        sf[nf][e] = xv;
        nm[e] = fmaxf(nm[e], xv);
      }
#pragma unroll
    for (int m = 1; m <= 8; m <<= 1)
#pragma unroll
      for (int e = 0; e < 8; ++e) nm[e] = fmaxf(nm[e], __shfl_xor(nm[e], m, 32));

    float alpha[8];
#pragma unroll
    for (int e = 0; e < 8; ++e) {
      float mn = fmaxf(mrow[e], nm[e]);
      alpha[e] = __expf(mrow[e] - mn);
      mrow[e] = mn;
    }
    float rs[8];
#pragma unroll
    for (int e = 0; e < 8; ++e) rs[e] = 0.f;
#pragma unroll
    for (int nf = 0; nf < 4; ++nf)
#pragma unroll
      for (int e = 0; e < 8; ++e) {
        float pv = __expf(sf[nf][e] - mrow[e]);
        sf[nf][e] = pv;
        rs[e] += pv;
      }
#pragma unroll
    for (int m = 1; m <= 8; m <<= 1)
#pragma unroll
      for (int e = 0; e < 8; ++e) rs[e] += __shfl_xor(rs[e], m, 32);
#pragma unroll
    for (int e = 0; e < 8; ++e) lrow[e] = lrow[e] * alpha[e] + rs[e];
#pragma unroll
    for (int dn = 0; dn < 4; ++dn)
#pragma unroll
      for (int e = 0; e < 8; ++e) oacc[dn][e] *= alpha[e];

    // P -> per-wave LDS (same-wave LDS ops are in-order), re-layout as A frag
    unsigned short* pw = (unsigned short*)(Ps + wave * 16 * 33);
#pragma unroll
    for (int nf = 0; nf < 4; ++nf)
#pragma unroll
      for (int e = 0; e < 8; ++e)
        pw[(e + 8 * hi) * 66 + nf * 16 + l16] = f2bf(sf[nf][e]);

    const unsigned* pu = Ps + wave * 16 * 33;
    const unsigned* vtu = (const unsigned*)VT;
#pragma unroll
    for (int ks = 0; ks < 2; ++ks) {         // O += P * V
      FragB a;
#pragma unroll
      for (int e = 0; e < 8; ++e)
        a.u[e] = pu[l16 * 33 + ks * 16 + hi * 8 + e];
#pragma unroll
      for (int dn = 0; dn < 4; ++dn) {
        FragB bv;
#pragma unroll
        for (int e = 0; e < 8; ++e)
          bv.u[e] = vtu[(dn * 16 + l16) * 33 + ks * 16 + hi * 8 + e];
        oacc[dn] = WMMA_BF16(a.v, bv.v, oacc[dn]);
      }
    }
  }
#pragma unroll
  for (int dn = 0; dn < 4; ++dn)
#pragma unroll
    for (int e = 0; e < 8; ++e) {
      int qr = qbase + e + 8 * hi;
      float o = oacc[dn][e] / lrow[e];
      outp[(size_t)(b * T + qr) * 1024 + h * 64 + dn * 16 + l16] = f2bf(o);
    }
}

// ---------------------------------------------------------------- SwiGLU
__global__ void silu_mul_kernel(const unsigned short* __restrict__ g,
                                const unsigned short* __restrict__ u,
                                unsigned short* __restrict__ s, int n) {
  int i = blockIdx.x * blockDim.x + threadIdx.x;
  if (i >= n) return;
  float xg = bf2f(g[i]);
  float xu = bf2f(u[i]);
  s[i] = f2bf(xg / (1.f + __expf(-xg)) * xu);
}

// ---------------------------------------------------------------- host
static inline unsigned cdiv(unsigned a, unsigned b) { return (a + b - 1) / b; }

extern "C" void kernel_launch(void* const* d_in, const int* in_sizes, int n_in,
                              void* d_out, int out_size, void* d_ws, size_t ws_size,
                              hipStream_t stream) {
  (void)in_sizes; (void)n_in; (void)out_size; (void)ws_size;
  const float* x = (const float*)d_in[0];
  const float* cores[5][5];
  for (int wgt = 0; wgt < 5; ++wgt)
    for (int s = 0; s < 5; ++s)
      cores[wgt][s] = (const float*)d_in[1 + wgt * 5 + s];
  const float* n1w = (const float*)d_in[26];
  const float* n2w = (const float*)d_in[27];

  char* ws = (char*)d_ws;
  float* tmpA = (float*)(ws + 0);
  float* tmpB = (float*)(ws + 16777216);
  unsigned short* Wqkv = (unsigned short*)(ws + 33554432);
  unsigned short* Wo   = (unsigned short*)(ws + 39845888);
  unsigned short* Wg   = (unsigned short*)(ws + 41943040);
  unsigned short* Wu   = (unsigned short*)(ws + 46137344);
  unsigned short* Wd   = (unsigned short*)(ws + 50331648);
  unsigned short* Hbuf = (unsigned short*)(ws + 54525952);
  unsigned short* QKV  = (unsigned short*)(ws + 62914560);
  float* X1 = tmpA;                       // reuse after weight build
  unsigned short* Gbuf = (unsigned short*)tmpB;
  unsigned short* Ubuf = QKV;             // reuse after attention
  unsigned short* ATT  = Hbuf;            // h -> attn_out -> h2, sequential
  float* OUT = (float*)d_out;

  // ---- build bf16 weights from TT cores ----
  struct Spec { int p, q, out_f, in_f; unsigned short* W; int wi; };
  const Spec specs[5] = {
    {4, 5, 3072, 1024, Wqkv, 0},
    {4, 4, 1024, 1024, Wo,   1},
    {4, 5, 2048, 1024, Wg,   2},
    {4, 5, 2048, 1024, Wu,   3},
    {5, 4, 1024, 2048, Wd,   4},
  };
  for (int i = 0; i < 5; ++i) {
    const Spec& sp = specs[i];
    const float* const* c = cores[sp.wi];
    int pq = sp.p * sp.q;
    int M = pq, tot;
    tot = M * pq * 16;
    tt_contract<<<cdiv(tot, 256), 256, 0, stream>>>(c[0], c[1], tmpA, M, pq, 16);
    M *= pq; tot = M * pq * 16;
    tt_contract<<<cdiv(tot, 256), 256, 0, stream>>>(tmpA, c[2], tmpB, M, pq, 16);
    M *= pq; tot = M * pq * 16;
    tt_contract<<<cdiv(tot, 256), 256, 0, stream>>>(tmpB, c[3], tmpA, M, pq, 16);
    M *= pq; tot = M * pq;
    tt_contract<<<cdiv(tot, 256), 256, 0, stream>>>(tmpA, c[4], tmpB, M, pq, 1);
    tot = sp.out_f * sp.in_f;
    tt_materialize<<<cdiv(tot, 256), 256, 0, stream>>>(tmpB, sp.W, sp.p, sp.q, sp.out_f, sp.in_f);
  }

  // ---- transformer block ----
  rmsnorm_kernel<<<4096, 256, 0, stream>>>(x, n1w, Hbuf, 1024);
  gemm_nt_kernel<1, 0><<<dim3(3072 / 128, 4096 / 128), 256, 0, stream>>>(
      Hbuf, Wqkv, nullptr, QKV, 4096, 3072, 1024);
  attention_kernel<<<512, 256, 0, stream>>>(QKV, ATT, 2048);
  gemm_nt_kernel<0, 1><<<dim3(1024 / 128, 4096 / 128), 256, 0, stream>>>(
      ATT, Wo, x, X1, 4096, 1024, 1024);
  rmsnorm_kernel<<<4096, 256, 0, stream>>>(X1, n2w, Hbuf, 1024);
  gemm_nt_kernel<1, 0><<<dim3(2048 / 128, 4096 / 128), 256, 0, stream>>>(
      Hbuf, Wg, nullptr, Gbuf, 4096, 2048, 1024);
  gemm_nt_kernel<1, 0><<<dim3(2048 / 128, 4096 / 128), 256, 0, stream>>>(
      Hbuf, Wu, nullptr, Ubuf, 4096, 2048, 1024);
  silu_mul_kernel<<<cdiv(4096u * 2048u, 256), 256, 0, stream>>>(
      Gbuf, Ubuf, Gbuf, 4096 * 2048);
  gemm_nt_kernel<0, 1><<<dim3(1024 / 128, 4096 / 128), 256, 0, stream>>>(
      Gbuf, Wd, X1, OUT, 4096, 1024, 2048);
}
